// Spectral_Attention_Structured_Mesh_2D_87436944212221
// MI455X (gfx1250) — matrix operand order repaired and checked by reading, not compiled, not run on hardware
//
#include <hip/hip_runtime.h>
#include <hip/hip_bf16.h>

typedef _Float16 f16;
typedef _Float16 h8  __attribute__((ext_vector_type(8)));
typedef _Float16 h16 __attribute__((ext_vector_type(16)));
typedef float    f8v __attribute__((ext_vector_type(8)));

#define TWO_PI_OVER_64 0.0981747704246810387f

constexpr int    Bm = 16, Ll = 1024, Wf = 33;
constexpr int    Mbig = 65536;   // B * H * W
constexpr int    Msm  = 16384;   // B * (H/2) * (W/2)
constexpr size_t FFT_ELEMS = (size_t)Bm * 64 * Wf * 256;   // 8,650,752
constexpr size_t XDWT_ELEMS = (size_t)Bm * 34 * 34 * 256;  // haloed NHWC

// ---------------------------------------------------------------- fragments
__device__ __forceinline__ h16 frag_ld(const f16* p0, const f16* p1) {
  h8 lo = *(const h8*)p0;
  h8 hi = *(const h8*)p1;
  h16 r;
#pragma unroll
  for (int i = 0; i < 8; ++i) { r[i] = lo[i]; r[i + 8] = hi[i]; }
  return r;
}
__device__ __forceinline__ float warp_sum(float v) {
#pragma unroll
  for (int m = 16; m; m >>= 1) v += __shfl_xor(v, m, 32);
  return v;
}
__device__ __forceinline__ float elu1(float x) {  // elu(x) + 1
  return x > 0.0f ? x + 1.0f : __expf(x);
}

// ---------------------------------------------------------------- WMMA GEMM
// A: [M][K] f16 row-major.  Bt: [N][K] f16 (== torch-style weight w[n][k]).
// MODE 0: out = acc + bias            (qkv)
// MODE 1: out = acc*scale + shift     (reduce conv + folded BN)
// MODE 2: auxh[m*512+256+n] = f16(acc + bias)   (proj -> x_wave half of merge A)
// MODE 3: s = sigmoid(acc+bias); out = s*xf + (1-s)*xw from auxh (merge+blend)
template <int MODE>
__global__ void gemm_wmma_k(const f16* __restrict__ A, const f16* __restrict__ Bt,
                            float* __restrict__ out, const float* __restrict__ bias,
                            const float* __restrict__ scale, const float* __restrict__ shift,
                            f16* __restrict__ auxh, int M, int N, int K) {
  const int wave   = (blockIdx.x * blockDim.x + threadIdx.x) >> 5;
  const int ntiles = N >> 4;
  const int total  = (M >> 4) * ntiles;
  if (wave >= total) return;
  const int tm = wave / ntiles;
  const int tn = wave - tm * ntiles;
  const int lane = threadIdx.x & 31;
  const int hlf = lane >> 4, l15 = lane & 15;
  const f16* ap = A + (size_t)(tm * 16 + l15) * K + 8 * hlf;
  const f16* bp = Bt + (size_t)(tn * 16 + l15) * K + 8 * hlf;
  // software pipeline: loads for step k+32 issued before the WMMA for step k
  h16 a = frag_ld(ap, ap + 16);
  h16 b = frag_ld(bp, bp + 16);
  f8v acc = {0.f, 0.f, 0.f, 0.f, 0.f, 0.f, 0.f, 0.f};
  for (int k0 = 32; k0 < K; k0 += 32) {
    h16 an = frag_ld(ap + k0, ap + k0 + 16);
    h16 bn = frag_ld(bp + k0, bp + k0 + 16);
    __builtin_prefetch(ap + k0 + 32, 0, 1);
    __builtin_prefetch(bp + k0 + 32, 0, 1);
    acc = __builtin_amdgcn_wmma_f32_16x16x32_f16(false, a, false, b, (short)0, acc,
                                                 false, false);
    a = an;
    b = bn;
  }
  acc = __builtin_amdgcn_wmma_f32_16x16x32_f16(false, a, false, b, (short)0, acc,
                                               false, false);
  const int   n  = tn * 16 + l15;
  const float bn_ = (MODE == 1) ? 0.0f : bias[n];
  const float sc  = (MODE == 1) ? scale[n] : 0.0f;
  const float sh  = (MODE == 1) ? shift[n] : 0.0f;
#pragma unroll
  for (int v = 0; v < 8; ++v) {
    const int   m   = tm * 16 + v + 8 * hlf;
    const float val = acc[v];
    if (MODE == 0) {
      out[(size_t)m * N + n] = val + bn_;
    } else if (MODE == 1) {
      out[(size_t)m * N + n] = val * sc + sh;
    } else if (MODE == 2) {
      auxh[(size_t)m * 512 + 256 + n] = (f16)(val + bn_);
    } else {
      const float s  = 1.0f / (1.0f + __expf(-(val + bn_)));
      const float xf = (float)auxh[(size_t)m * 512 + n];
      const float xw = (float)auxh[(size_t)m * 512 + 256 + n];
      out[(size_t)m * N + n] = s * xf + (1.0f - s) * xw;
    }
  }
}

// ------------------------------------------------ implicit-GEMM 3x3 conv (SAME)
// X: x_dwt NHWC with 1px zero halo (B,34,34,256) f16 -> no bounds checks.
// Wp: [tap][O][I] f16.  out: (B*L,256) f32 with folded BN.
__global__ void conv3x3_wmma_k(const f16* __restrict__ X, const f16* __restrict__ Wp,
                               float* __restrict__ out, const float* __restrict__ scale,
                               const float* __restrict__ shift) {
  const int wave = (blockIdx.x * blockDim.x + threadIdx.x) >> 5;
  if (wave >= 16384) return;
  const int tm = wave >> 4, tn = wave & 15;
  const int lane = threadIdx.x & 31;
  const int hlf = lane >> 4, l15 = lane & 15;
  const int m = tm * 16 + l15;
  const int b = m >> 10, rem = m & 1023, y = rem >> 5, x = rem & 31;
  // halo coords: interior pixel (y,x) lives at (y+1, x+1)
  const f16* acenter = X + ((size_t)((b * 34 + y + 1) * 34 + (x + 1)) << 8) + 8 * hlf;
  const f16* brow = Wp + (size_t)(tn * 16 + l15) * 256 + 8 * hlf;
  f8v acc = {0.f, 0.f, 0.f, 0.f, 0.f, 0.f, 0.f, 0.f};
#pragma unroll
  for (int tap = 0; tap < 9; ++tap) {
    const int dy = tap / 3 - 1, dx = tap - (tap / 3) * 3 - 1;
    const f16* arow = acenter + ((size_t)(dy * 34 + dx) << 8);
    const f16* wrow = brow + (size_t)tap * 65536;
#pragma unroll
    for (int k0 = 0; k0 < 256; k0 += 32) {
      h16 a  = frag_ld(arow + k0, arow + k0 + 16);
      h16 bf = frag_ld(wrow + k0, wrow + k0 + 16);
      acc = __builtin_amdgcn_wmma_f32_16x16x32_f16(false, a, false, bf, (short)0, acc,
                                                   false, false);
    }
  }
  const int   n  = tn * 16 + l15;
  const float sc = scale[n], sh = shift[n];
#pragma unroll
  for (int v = 0; v < 8; ++v) {
    const int mo = tm * 16 + v + 8 * hlf;
    out[(size_t)mo * 256 + n] = acc[v] * sc + sh;
  }
}

// ---------------------------------------------------------------- FFT branch
// one block == one (b,y,kx); twiddles shared through LDS
__global__ void rfft_rows_k(const float* __restrict__ x, float* __restrict__ oR,
                            float* __restrict__ oI) {
  __shared__ float tc[64], ts[64];
  const size_t t    = (size_t)blockIdx.x * 256 + threadIdx.x;
  const int    kx   = (int)((t >> 8) % 33);
  const size_t rest = (t >> 8) / 33;  // b*64 + y
  if (threadIdx.x < 64) {
    float s, c;
    __sincosf(-TWO_PI_OVER_64 * (float)((kx * threadIdx.x) & 63), &s, &c);
    tc[threadIdx.x] = c;
    ts[threadIdx.x] = s;
  }
  __syncthreads();
  const float* row = x + (rest << 14) + (t & 255);
  float aR = 0.f, aI = 0.f;
  for (int j = 0; j < 64; ++j) {
    const float v = row[(size_t)j << 8];
    aR += v * tc[j];
    aI += v * ts[j];
  }
  oR[t] = aR * 0.125f;
  oI[t] = aI * 0.125f;
}

__global__ void cfft_cols_k(const float* __restrict__ iR, const float* __restrict__ iI,
                            float* __restrict__ oR, float* __restrict__ oI, float sgn) {
  __shared__ float tc[64], ts[64];
  const size_t t  = (size_t)blockIdx.x * 256 + threadIdx.x;
  const int    kx = (int)((t >> 8) % 33);
  const size_t by = (t >> 8) / 33;
  const int    ky = (int)(by & 63);
  const int    b  = (int)(by >> 6);
  if (threadIdx.x < 64) {
    float s, c;
    __sincosf(sgn * TWO_PI_OVER_64 * (float)((ky * threadIdx.x) & 63), &s, &c);
    tc[threadIdx.x] = c;
    ts[threadIdx.x] = s;
  }
  __syncthreads();
  const size_t base = (((size_t)b * 64) * 33 + kx) * 256 + (t & 255);  // y = 0
  const size_t ystr = (size_t)33 * 256;
  float aR = 0.f, aI = 0.f;
  for (int y = 0; y < 64; ++y) {
    const float r  = iR[base + y * ystr];
    const float im = iI[base + y * ystr];
    aR += r * tc[y] - im * ts[y];
    aI += r * ts[y] + im * tc[y];
  }
  oR[t] = aR * 0.125f;
  oI[t] = aI * 0.125f;
}

// block-diagonal complex 2-layer MLP, in place on (B,H,33,256)
__global__ void spec_mlp_k(float* __restrict__ R, float* __restrict__ I,
                           const float* __restrict__ w1, const float* __restrict__ b1,
                           const float* __restrict__ w2, const float* __restrict__ b2) {
  __shared__ float xr[256], xi[256], yr[256], yi[256];
  const size_t base = (size_t)blockIdx.x << 8;
  const int tid = threadIdx.x;
  xr[tid] = R[base + tid];
  xi[tid] = I[base + tid];
  __syncthreads();
  const int nb = tid >> 5, o = tid & 31, xb = nb << 5;
  const float* w1r = w1 + (size_t)nb * 1024;
  const float* w1i = w1 + 8192 + (size_t)nb * 1024;
  float ar = b1[nb * 32 + o], ai = b1[256 + nb * 32 + o];
  for (int i = 0; i < 32; ++i) {
    const float wr = w1r[i * 32 + o], wi = w1i[i * 32 + o];
    ar += xr[xb + i] * wr - xi[xb + i] * wi;
    ai += xi[xb + i] * wr + xr[xb + i] * wi;
  }
  yr[tid] = fmaxf(ar, 0.f);
  yi[tid] = fmaxf(ai, 0.f);
  __syncthreads();
  const float* w2r = w2 + (size_t)nb * 1024;
  const float* w2i = w2 + 8192 + (size_t)nb * 1024;
  float pr = b2[nb * 32 + o], pi = b2[256 + nb * 32 + o];
  for (int i = 0; i < 32; ++i) {
    const float wr = w2r[i * 32 + o], wi = w2i[i * 32 + o];
    pr += yr[xb + i] * wr - yi[xb + i] * wi;
    pi += yi[xb + i] * wr + yr[xb + i] * wi;
  }
  R[base + tid] = pr;
  I[base + tid] = pi;
}

// Hermitian inverse row transform + residual; writes f16 x_fft into merge A cols [0,256)
__global__ void irfft_rows_k(const float* __restrict__ R, const float* __restrict__ I,
                             const float* __restrict__ xori, f16* __restrict__ amerge) {
  __shared__ float tc[32], ts[32];
  const size_t t  = (size_t)blockIdx.x * 256 + threadIdx.x;
  const size_t m  = t >> 8;       // b*4096 + y*64 + x ; one block per m
  const int    xx = (int)(m & 63);
  if (threadIdx.x < 32) {
    float s, c;
    __sincosf(TWO_PI_OVER_64 * (float)((threadIdx.x * xx) & 63), &s, &c);
    tc[threadIdx.x] = c;
    ts[threadIdx.x] = s;
  }
  __syncthreads();
  const size_t by = m >> 6;  // b*64 + y
  const float* Rb = R + by * 33 * 256 + (t & 255);
  const float* Ib = I + by * 33 * 256 + (t & 255);
  float acc = Rb[0] + ((xx & 1) ? -Rb[32 * 256] : Rb[32 * 256]);
  for (int k = 1; k < 32; ++k)
    acc += 2.0f * (Rb[(size_t)k << 8] * tc[k] - Ib[(size_t)k << 8] * ts[k]);
  acc = acc * 0.125f + xori[t];
  amerge[m * 512 + (t & 255)] = (f16)acc;
}

// ---------------------------------------------------------------- prep kernels
__global__ void cvt_x_k(const float* __restrict__ x, f16* __restrict__ xh,
                        f16* __restrict__ aproj) {
  const size_t t = (size_t)blockIdx.x * blockDim.x + threadIdx.x;
  if (t >= (size_t)Mbig * 256) return;
  const float v = x[t];
  xh[t] = (f16)v;
  aproj[(t >> 8) * 320 + (t & 255)] = (f16)v;
}
__global__ void cvt_f16_k(const float* __restrict__ s, f16* __restrict__ d, int n) {
  const int t = blockIdx.x * blockDim.x + threadIdx.x;
  if (t < n) d[t] = (f16)s[t];
}
__global__ void pack_filt_k(const float* __restrict__ w, f16* __restrict__ o) {
  const int t = blockIdx.x * blockDim.x + threadIdx.x;
  if (t >= 9 * 65536) return;
  const int i = t & 255, oc = (t >> 8) & 255, tap = t >> 16;
  o[t] = (f16)w[(size_t)((oc << 8) + i) * 9 + tap];
}
__global__ void fold_bn_k(const float* g, const float* bt, const float* mean,
                          const float* var, const float* bias, float* scale,
                          float* shift, int n) {
  const int t = blockIdx.x * blockDim.x + threadIdx.x;
  if (t >= n) return;
  const float sc = g[t] * rsqrtf(var[t] + 1e-5f);
  scale[t] = sc;
  shift[t] = (bias[t] - mean[t]) * sc + bt[t];
}
__global__ void zero_h8_k(f16* __restrict__ p, size_t n8) {  // clear in 16B chunks
  const size_t t = (size_t)blockIdx.x * blockDim.x + threadIdx.x;
  if (t >= n8) return;
  h8 z;
#pragma unroll
  for (int i = 0; i < 8; ++i) z[i] = (f16)0.0f;
  ((h8*)p)[t] = z;
}

// ---------------------------------------------------------------- wavelet ops
__global__ void dwt_k(const float* __restrict__ xr, f16* __restrict__ o) {
  const int t = blockIdx.x * blockDim.x + threadIdx.x;
  if (t >= 16 * 32 * 32 * 64) return;
  const int cc = t & 63, x = (t >> 6) & 31, y = (t >> 11) & 31, b = t >> 16;
  const float* p = xr + (((size_t)(b * 64 + 2 * y) * 64 + 2 * x) << 6) + cc;
  const float a = p[0], bb = p[64], c = p[4096], d = p[4096 + 64];
  // write into haloed (B,34,34,256) buffer at (y+1, x+1)
  f16* q = o + (((size_t)((b * 34 + y + 1) * 34 + x + 1)) << 8) + cc;
  q[0]   = (f16)((a + bb + c + d) * 0.5f);
  q[64]  = (f16)((a + bb - c - d) * 0.5f);
  q[128] = (f16)((a - bb + c - d) * 0.5f);
  q[192] = (f16)((a - bb - c + d) * 0.5f);
}

__global__ void ln_k(const float* __restrict__ kv, const float* __restrict__ g,
                     const float* __restrict__ bt, f16* __restrict__ o) {
  const int row  = blockIdx.x * 8 + (threadIdx.x >> 5);
  const int lane = threadIdx.x & 31;
  const float* p = kv + (size_t)row * 256;
  float vals[8], s = 0.f, s2 = 0.f;
#pragma unroll
  for (int j = 0; j < 8; ++j) {
    const float v = p[lane + (j << 5)];
    vals[j] = v; s += v; s2 += v * v;
  }
  s  = warp_sum(s);
  s2 = warp_sum(s2);
  const float mu  = s * (1.0f / 256.0f);
  const float inv = rsqrtf(s2 * (1.0f / 256.0f) - mu * mu + 1e-5f);
#pragma unroll
  for (int j = 0; j < 8; ++j) {
    const int c = lane + (j << 5);
    o[(size_t)row * 256 + c] = (f16)((vals[j] - mu) * inv * g[c] + bt[c]);
  }
}

__global__ void kv_k(const float* __restrict__ qkv, float* __restrict__ KV,
                     float* __restrict__ Ks) {
  const int bh = blockIdx.x;          // B*NB = 128
  const int b = bh >> 3, h = bh & 7;
  const int d = threadIdx.x & 31, mm = threadIdx.x >> 5;  // 1024 threads
  const float* base = qkv + (size_t)b * Ll * 768 + h * 32;
  float acc = 0.f, ks = 0.f;
  for (int s = 0; s < Ll; ++s) {
    const float kf = elu1(base[(size_t)s * 768 + 256 + d]);
    const float vv = base[(size_t)s * 768 + 512 + mm];
    acc += kf * vv;
    ks += kf;
  }
  KV[((size_t)bh * 32 + mm) * 32 + d] = acc;
  if (mm == 0) Ks[(size_t)bh * 32 + d] = ks;
}

__global__ void attn_k(const float* __restrict__ qkv, const float* __restrict__ KV,
                       const float* __restrict__ Ks, float* __restrict__ attn) {
  const int unit = blockIdx.x * 8 + (threadIdx.x >> 5);  // B*L*NB = 131072
  const int lane = threadIdx.x & 31;
  const int h  = unit & 7;
  const int bl = unit >> 3;
  const int b  = bl >> 10;
  const int bh = b * 8 + h;
  const float q     = elu1(qkv[(size_t)bl * 768 + h * 32 + lane]);
  const float denom = warp_sum(q * Ks[(size_t)bh * 32 + lane]) + 1e-6f;
  const float* kvp  = KV + ((size_t)bh * 32 + lane) * 32;  // m = lane
  float acc = 0.f;
#pragma unroll
  for (int d2 = 0; d2 < 32; ++d2) acc += __shfl(q, d2, 32) * kvp[d2];
  attn[(size_t)bl * 256 + h * 32 + lane] = acc / denom;
}

__global__ void idwt_k(const float* __restrict__ attn, f16* __restrict__ aproj) {
  const int t = blockIdx.x * blockDim.x + threadIdx.x;
  if (t >= 16 * 32 * 32 * 64) return;
  const int cc = t & 63, x = (t >> 6) & 31, y = (t >> 11) & 31, b = t >> 16;
  const float* p = attn + (((size_t)(b * 32 + y) * 32 + x) << 8) + cc;
  const float ll = p[0], lh = p[64], hl = p[128], hh = p[192];
  const float a = (ll + lh + hl + hh) * 0.5f, bb = (ll + lh - hl - hh) * 0.5f;
  const float c = (ll - lh + hl - hh) * 0.5f, d = (ll - lh - hl + hh) * 0.5f;
  const size_t m00 = (size_t)b * 4096 + (2 * y) * 64 + 2 * x;
  f16* q = aproj + m00 * 320 + 256 + cc;
  q[0]        = (f16)a;
  q[320]      = (f16)bb;
  q[64 * 320] = (f16)c;
  q[65 * 320] = (f16)d;
}

// ---------------------------------------------------------------- host driver
extern "C" void kernel_launch(void* const* d_in, const int* in_sizes, int n_in,
                              void* d_out, int out_size, void* d_ws, size_t ws_size,
                              hipStream_t stream) {
  (void)in_sizes; (void)n_in; (void)out_size; (void)ws_size;
  const float* x        = (const float*)d_in[0];
  const float* w1       = (const float*)d_in[1];
  const float* b1       = (const float*)d_in[2];
  const float* w2       = (const float*)d_in[3];
  const float* b2       = (const float*)d_in[4];
  const float* reduce_w = (const float*)d_in[5];
  const float* reduce_b = (const float*)d_in[6];
  const float* rbn_g    = (const float*)d_in[7];
  const float* rbn_b    = (const float*)d_in[8];
  const float* rbn_m    = (const float*)d_in[9];
  const float* rbn_v    = (const float*)d_in[10];
  const float* filt_w   = (const float*)d_in[11];
  const float* filt_b   = (const float*)d_in[12];
  const float* fbn_g    = (const float*)d_in[13];
  const float* fbn_b    = (const float*)d_in[14];
  const float* fbn_m    = (const float*)d_in[15];
  const float* fbn_v    = (const float*)d_in[16];
  const float* ln_g     = (const float*)d_in[17];
  const float* ln_b     = (const float*)d_in[18];
  const float* qkv_w    = (const float*)d_in[19];
  const float* qkv_b    = (const float*)d_in[20];
  const float* proj_w   = (const float*)d_in[21];
  const float* proj_b   = (const float*)d_in[22];
  const float* merge_w  = (const float*)d_in[23];
  const float* merge_b  = (const float*)d_in[24];
  float* out = (float*)d_out;

  size_t off = 0;
  auto alloc = [&](size_t bytes) -> void* {
    void* p = (char*)d_ws + off;
    off += (bytes + 255) & ~(size_t)255;
    return p;
  };

  f16*   xh       = (f16*)alloc((size_t)Mbig * 256 * 2);
  f16*   aproj    = (f16*)alloc((size_t)Mbig * 320 * 2);
  f16*   amerge   = (f16*)alloc((size_t)Mbig * 512 * 2);
  f16*   qkvw_h   = (f16*)alloc((size_t)768 * 256 * 2);
  f16*   projw_h  = (f16*)alloc((size_t)256 * 320 * 2);
  f16*   mergew_h = (f16*)alloc((size_t)256 * 512 * 2);
  f16*   redw_h   = (f16*)alloc((size_t)64 * 256 * 2);
  f16*   filtw_h  = (f16*)alloc((size_t)9 * 256 * 256 * 2);
  float* rsc      = (float*)alloc(64 * 4);
  float* rsh      = (float*)alloc(64 * 4);
  float* fsc      = (float*)alloc(256 * 4);
  float* fsh      = (float*)alloc(256 * 4);

  // FFT scratch / wavelet scratch share one region (branches run sequentially)
  char*  scr   = (char*)alloc(4 * FFT_ELEMS * sizeof(float));
  float* fftR  = (float*)scr;
  float* fftI  = fftR + FFT_ELEMS;
  float* fft2R = fftI + FFT_ELEMS;
  float* fft2I = fft2R + FFT_ELEMS;
  char* wp = scr;
  float* xred = (float*)wp;  wp += (size_t)Mbig * 64 * 4;
  f16*   xdwt = (f16*)wp;    wp += XDWT_ELEMS * 2;          // haloed
  float* kvb  = (float*)wp;  wp += (size_t)Msm * 256 * 4;
  f16*   lnh  = (f16*)wp;    wp += (size_t)Msm * 256 * 2;
  float* qkvb = (float*)wp;  wp += (size_t)Msm * 768 * 4;
  float* KVb  = (float*)wp;  wp += (size_t)128 * 32 * 32 * 4;
  float* Ksum = (float*)wp;  wp += (size_t)128 * 32 * 4;
  float* attn = (float*)wp;  wp += (size_t)Msm * 256 * 4;

  const int TB = 256;

  // --- prep: f16 conversions, weight packing, BN folds
  {
    size_t n = (size_t)Mbig * 256;
    cvt_x_k<<<(unsigned)((n + TB - 1) / TB), TB, 0, stream>>>(x, xh, aproj);
  }
  cvt_f16_k<<<(768 * 256 + TB - 1) / TB, TB, 0, stream>>>(qkv_w, qkvw_h, 768 * 256);
  cvt_f16_k<<<(256 * 320 + TB - 1) / TB, TB, 0, stream>>>(proj_w, projw_h, 256 * 320);
  cvt_f16_k<<<(256 * 512 + TB - 1) / TB, TB, 0, stream>>>(merge_w, mergew_h, 256 * 512);
  cvt_f16_k<<<(64 * 256 + TB - 1) / TB, TB, 0, stream>>>(reduce_w, redw_h, 64 * 256);
  pack_filt_k<<<(9 * 65536 + TB - 1) / TB, TB, 0, stream>>>(filt_w, filtw_h);
  fold_bn_k<<<1, 64, 0, stream>>>(rbn_g, rbn_b, rbn_m, rbn_v, reduce_b, rsc, rsh, 64);
  fold_bn_k<<<1, 256, 0, stream>>>(fbn_g, fbn_b, fbn_m, fbn_v, filt_b, fsc, fsh, 256);

  // --- FFT branch
  {
    const unsigned nb = (unsigned)(FFT_ELEMS / TB);  // exact multiple
    rfft_rows_k<<<nb, TB, 0, stream>>>(x, fftR, fftI);
    cfft_cols_k<<<nb, TB, 0, stream>>>(fftR, fftI, fft2R, fft2I, -1.0f);
    spec_mlp_k<<<16 * 64 * 33, 256, 0, stream>>>(fft2R, fft2I, w1, b1, w2, b2);
    cfft_cols_k<<<nb, TB, 0, stream>>>(fft2R, fft2I, fftR, fftI, 1.0f);
    irfft_rows_k<<<Mbig, TB, 0, stream>>>(fftR, fftI, x, amerge);
  }

  // --- wavelet branch
  {  // reduce 1x1 conv + BN : (65536x256) @ (256->64)
    const int tiles = (Mbig / 16) * (64 / 16);
    gemm_wmma_k<1><<<(tiles * 32 + TB - 1) / TB, TB, 0, stream>>>(
        xh, redw_h, xred, nullptr, rsc, rsh, nullptr, Mbig, 64, 256);
  }
  zero_h8_k<<<(unsigned)((XDWT_ELEMS / 8 + TB - 1) / TB), TB, 0, stream>>>(xdwt,
                                                                           XDWT_ELEMS / 8);
  dwt_k<<<(16 * 32 * 32 * 64 + TB - 1) / TB, TB, 0, stream>>>(xred, xdwt);
  conv3x3_wmma_k<<<(16384 * 32 + TB - 1) / TB, TB, 0, stream>>>(xdwt, filtw_h, kvb, fsc, fsh);
  ln_k<<<Msm / 8, 256, 0, stream>>>(kvb, ln_g, ln_b, lnh);
  {  // qkv : (16384x256) @ (256->768)
    const int tiles = (Msm / 16) * (768 / 16);
    gemm_wmma_k<0><<<(tiles * 32 + TB - 1) / TB, TB, 0, stream>>>(
        lnh, qkvw_h, qkvb, qkv_b, nullptr, nullptr, nullptr, Msm, 768, 256);
  }
  kv_k<<<128, 1024, 0, stream>>>(qkvb, KVb, Ksum);
  attn_k<<<131072 / 8, 256, 0, stream>>>(qkvb, KVb, Ksum, attn);
  idwt_k<<<(16 * 32 * 32 * 64 + TB - 1) / TB, TB, 0, stream>>>(attn, aproj);
  {  // proj : (65536x320) @ (320->256) -> f16 into merge A cols [256,512)
    const int tiles = (Mbig / 16) * (256 / 16);
    gemm_wmma_k<2><<<(tiles * 32 + TB - 1) / TB, TB, 0, stream>>>(
        aproj, projw_h, nullptr, proj_b, nullptr, nullptr, amerge, Mbig, 256, 320);
  }
  {  // merge + sigmoid + blend : (65536x512) @ (512->256) -> d_out
    const int tiles = (Mbig / 16) * (256 / 16);
    gemm_wmma_k<3><<<(tiles * 32 + TB - 1) / TB, TB, 0, stream>>>(
        amerge, mergew_h, out, merge_b, nullptr, nullptr, amerge, Mbig, 256, 512);
  }
}
